// QGroupLinear_63161789055351
// MI455X (gfx1250) — matrix operand order repaired and checked by reading, not compiled, not run on hardware
//
#include <hip/hip_runtime.h>

typedef __attribute__((ext_vector_type(16))) _Float16 v16h;
typedef __attribute__((ext_vector_type(8)))  _Float16 h8;
typedef __attribute__((ext_vector_type(4)))  _Float16 h4;
typedef __attribute__((ext_vector_type(8)))  float    v8f;
typedef __attribute__((ext_vector_type(4)))  float    f32x4;
typedef __attribute__((ext_vector_type(4)))  int      i32x4;

#define BM 128
#define BN 128
#define BK 128              // == GROUP_SIZE: one weight scale per row per K-tile
#define LDA (BK + 8)        // padded LDS row stride (halves); keeps 16B alignment
#define LDB (BK + 8)

union Frag { v16h v; h8 h[2]; };

__device__ __forceinline__
void compute_tile(const _Float16* __restrict__ As,
                  const _Float16* __restrict__ Bs,
                  v8f acc[4][2],
                  int wm, int wn, int lane15, int laneHi)
{
    #pragma unroll
    for (int kk = 0; kk < 4; ++kk) {
        const int kOff = kk * 32;

        // A fragments: 16x32 f16 (ISA layout):
        // lane<16: K={0..7},{16..23}; lane>=16: K={8..15},{24..31}
        Frag a[4];
        #pragma unroll
        for (int i = 0; i < 4; ++i) {
            const int row = wm * 64 + i * 16 + lane15;
            const int k0  = kOff + (laneHi ? 8 : 0);
            const _Float16* p = &As[row * LDA + k0];
            a[i].h[0] = *(const h8*)(p);
            a[i].h[1] = *(const h8*)(p + 16);
        }

        // B fragments: 32x16 f16 (ISA layout):
        // lane<16: K={0..7},{16..23}; lane>=16: K={8..15},{24..31}; N=lane&15
        Frag b[2];
        #pragma unroll
        for (int j = 0; j < 2; ++j) {
            const int n  = wn * 32 + j * 16 + lane15;
            const int k0 = kOff + (laneHi ? 8 : 0);
            const _Float16* p = &Bs[n * LDB + k0];
            b[j].h[0] = *(const h8*)(p);
            b[j].h[1] = *(const h8*)(p + 16);
        }

        #pragma unroll
        for (int i = 0; i < 4; ++i)
            #pragma unroll
            for (int j = 0; j < 2; ++j)
                acc[i][j] = __builtin_amdgcn_wmma_f32_16x16x32_f16(
                    /*neg_a=*/false, a[i].v,
                    /*neg_b=*/false, b[j].v,
                    /*c_mod=*/(short)0, acc[i][j],
                    /*reuse_a=*/false, /*reuse_b=*/false);
    }
}

__global__ __launch_bounds__(256)
void qgroup_linear_wmma(const float* __restrict__ X,   // [M, K] f32
                        const int*   __restrict__ QW,  // [N, K] int8-valued int32
                        const float* __restrict__ SW,  // [N, K/128] f32
                        const float* __restrict__ BIAS,// [N]
                        float*       __restrict__ Y,   // [M, N] f32
                        int M, int Ntot, int K)
{
    __shared__ _Float16 As[2][BM * LDA];   // 2 x 34 KB
    __shared__ _Float16 Bs[2][BN * LDB];   // 2 x 34 KB  (total 136 KB < 320 KB)

    const int tid  = threadIdx.x;       // 0..255
    const int lane = tid & 31;
    const int wave = tid >> 5;          // 0..7
    const int wm   = wave >> 2;         // 0..1  (M direction)
    const int wn   = wave & 3;          // 0..3  (N direction)

    const int blockN    = blockIdx.x * BN;
    const int blockM    = blockIdx.y * BM;
    const int numGroups = K / BK;
    const int kTiles    = K / BK;

    const int laneHi = lane >> 4;       // 0 or 1
    const int lane15 = lane & 15;

    // Each thread owns one fixed column-quad and 16 rows of the 128x128 tiles.
    const int col4 = tid & 31;          // float4 / int4 column index (0..31)
    const int row0 = tid >> 3 >> 2;     // == tid >> 5, 0..7

    v8f acc[4][2];
    #pragma unroll
    for (int i = 0; i < 4; ++i)
        #pragma unroll
        for (int j = 0; j < 2; ++j)
            acc[i][j] = (v8f){0.f,0.f,0.f,0.f,0.f,0.f,0.f,0.f};

    // In-flight staging registers (live across compute for latency hiding)
    f32x4 ra[16];
    i32x4 rb[16];
    float rs[16];

    // ---- Prologue: issue global loads for tile 0 ----
    {
        const int kBase = 0;
        #pragma unroll
        for (int it = 0; it < 16; ++it) {
            const int row = row0 + 8 * it;
            ra[it] = *(const f32x4*)(X  + (size_t)(blockM + row) * K + kBase + col4 * 4);
            rb[it] = *(const i32x4*)(QW + (size_t)(blockN + row) * K + kBase + col4 * 4);
            rs[it] = SW[(size_t)(blockN + row) * numGroups + 0];
        }
    }

    for (int kt = 0; kt < kTiles; ++kt) {
        const int p = kt & 1;

        // ---- Convert + store staged registers into LDS buffer p ----
        #pragma unroll
        for (int it = 0; it < 16; ++it) {
            const int row = row0 + 8 * it;
            h4 hv;
            hv.x = (_Float16)ra[it].x; hv.y = (_Float16)ra[it].y;
            hv.z = (_Float16)ra[it].z; hv.w = (_Float16)ra[it].w;
            *(h4*)(&As[p][row * LDA + col4 * 4]) = hv;

            const float s = rs[it];
            h4 hb;
            hb.x = (_Float16)((float)rb[it].x * s); hb.y = (_Float16)((float)rb[it].y * s);
            hb.z = (_Float16)((float)rb[it].z * s); hb.w = (_Float16)((float)rb[it].w * s);
            *(h4*)(&Bs[p][row * LDB + col4 * 4]) = hb;
        }

        __syncthreads();   // single barrier per K-tile (double-buffered)

        // ---- Issue global loads for tile kt+1 (overlap with compute below) ----
        if (kt + 1 < kTiles) {
            const int kBase = (kt + 1) * BK;
            #pragma unroll
            for (int it = 0; it < 16; ++it) {
                const int row = row0 + 8 * it;
                ra[it] = *(const f32x4*)(X  + (size_t)(blockM + row) * K + kBase + col4 * 4);
                rb[it] = *(const i32x4*)(QW + (size_t)(blockN + row) * K + kBase + col4 * 4);
                rs[it] = SW[(size_t)(blockN + row) * numGroups + (kt + 1)];
            }
        }

        // ---- Compute on buffer p while next tile's loads are in flight ----
        compute_tile(&As[p][0], &Bs[p][0], acc, wm, wn, lane15, laneHi);
    }

    // ---- Epilogue: C/D layout (VGPR r -> M row; lane&15 -> N col; lane>=16 -> M+8) ----
    #pragma unroll
    for (int j = 0; j < 2; ++j) {
        const int nCol = blockN + wn * 32 + j * 16 + lane15;
        const float bz = BIAS[nCol];
        #pragma unroll
        for (int i = 0; i < 4; ++i) {
            const int mBase = blockM + wm * 64 + i * 16 + (laneHi ? 8 : 0);
            #pragma unroll
            for (int r = 0; r < 8; ++r) {
                Y[(size_t)(mBase + r) * Ntot + nCol] = acc[i][j][r] + bz;
            }
        }
    }
}

extern "C" void kernel_launch(void* const* d_in, const int* in_sizes, int n_in,
                              void* d_out, int out_size, void* d_ws, size_t ws_size,
                              hipStream_t stream) {
    const float* X    = (const float*)d_in[0];
    const int*   QW   = (const int*)d_in[1];
    const float* SW   = (const float*)d_in[2];
    const float* BIAS = (const float*)d_in[3];
    float*       Y    = (float*)d_out;

    const int N         = in_sizes[3];              // 11008
    const int numGroups = in_sizes[2] / N;          // 32
    const int K         = numGroups * 128;          // 4096
    const int M         = in_sizes[0] / K;          // 4096 (= B*S)

    dim3 grid(N / BN, M / BM);                      // (86, 32)
    qgroup_linear_wmma<<<grid, 256, 0, stream>>>(X, QW, SW, BIAS, Y, M, N, K);
}
